// MoELayer_52750788329545
// MI455X (gfx1250) — compile-verified, matrix operand order checked by reading, not run on hardware
//
#include <hip/hip_runtime.h>

// ---------------- problem constants ----------------
constexpr int MD  = 1024;   // model dim
constexpr int DH  = 2048;   // expert hidden dim
constexpr int NE  = 8;      // experts
constexpr int TT  = 8192;   // tokens = B*N

// ---------------- types ----------------
typedef __bf16 bf16;
typedef bf16  v16bf __attribute__((ext_vector_type(16)));
typedef bf16  v8bf  __attribute__((ext_vector_type(8)));
typedef float v8f   __attribute__((ext_vector_type(8)));

union Frag16 { v16bf v; v8bf h[2]; };
union Acc8   { v8f v; float f[8]; };

constexpr int LDS_STRIDE = 40;                       // 32 + 8 pad halves (80B rows, 16B aligned)
constexpr unsigned A_BUF_B = 128 * LDS_STRIDE * 2;   // bytes per A buffer
constexpr unsigned B_BUF_B = 64  * LDS_STRIDE * 2;   // bytes per B buffer

// Load a 16x32 bf16 WMMA fragment from an LDS tile (row-major, stride LDS_STRIDE).
// ISA 7.12.2: lanes 0-15 hold K 0-7 / 16-23, lanes 16-31 hold K 8-15 / 24-31.
__device__ __forceinline__ v16bf ld_frag(const bf16* base, int row, int lmod, int k0) {
  const bf16* p = base + (row + lmod) * LDS_STRIDE;
  Frag16 f;
  f.h[0] = *(const v8bf*)(p + k0);
  f.h[1] = *(const v8bf*)(p + k0 + 16);
  return f.v;
}

__device__ __forceinline__ void async_cp16(unsigned lds_addr, unsigned goff, uint64_t gbase) {
  asm volatile("global_load_async_to_lds_b128 %0, %1, %2"
               :: "v"(lds_addr), "v"(goff), "s"(gbase) : "memory");
}
__device__ __forceinline__ void wait_async3() { asm volatile("s_wait_asynccnt 0x3" ::: "memory"); }
__device__ __forceinline__ void wait_async0() { asm volatile("s_wait_asynccnt 0x0" ::: "memory"); }

// ---------------- 0: zero workspace ----------------
__global__ void zero_kernel(float* __restrict__ h, size_t n,
                            int* __restrict__ cnt, float* __restrict__ gsum,
                            float* __restrict__ psum) {
  size_t i = (size_t)blockIdx.x * blockDim.x + threadIdx.x;
  size_t stride = (size_t)gridDim.x * blockDim.x;
  for (size_t j = i; j < n; j += stride) h[j] = 0.0f;
  if (i < NE) { cnt[i] = 0; gsum[i] = 0.0f; psum[i] = 0.0f; }
}

// ---------------- precision prep: f32 -> bf16 (elementwise, vectorized) ----------------
__global__ void cvt_bf16_kernel(const float* __restrict__ in, bf16* __restrict__ out, size_t n4) {
  size_t i = (size_t)blockIdx.x * blockDim.x + threadIdx.x;
  size_t stride = (size_t)gridDim.x * blockDim.x;
  for (size_t j = i; j < n4; j += stride) {
    float4 v = ((const float4*)in)[j];
    bf16* d = out + j * 4;
    d[0] = (bf16)v.x; d[1] = (bf16)v.y; d[2] = (bf16)v.z; d[3] = (bf16)v.w;
  }
}

// ---------------- precision prep: f32 [K][N] -> bf16 [N][K] (LDS-tiled transpose) ----------------
__global__ __launch_bounds__(256)
void transpose_cvt_kernel(const float* __restrict__ in, bf16* __restrict__ out,
                          int K, int N) {
  __shared__ float tile[32][33];
  size_t boff = (size_t)blockIdx.z * (size_t)K * N;
  const float* src = in + boff;
  bf16* dst = out + boff;
  int k0 = blockIdx.x * 32, n0 = blockIdx.y * 32;
  int tx = threadIdx.x & 31, ty = threadIdx.x >> 5;   // 32x8
  #pragma unroll
  for (int j = 0; j < 4; ++j)
    tile[ty + j * 8][tx] = src[(size_t)(k0 + ty + j * 8) * N + n0 + tx];
  __syncthreads();
  #pragma unroll
  for (int j = 0; j < 4; ++j)
    dst[(size_t)(n0 + ty + j * 8) * K + k0 + tx] = (bf16)tile[tx][ty + j * 8];
}

// ---------------- 1: gating (one wave per token) ----------------
__global__ __launch_bounds__(256)
void gate_kernel(const float* __restrict__ x, const float* __restrict__ noise,
                 const float* __restrict__ Wg, const float* __restrict__ bg,
                 const float* __restrict__ Wn, const float* __restrict__ bn,
                 int* __restrict__ cnt, float* __restrict__ gsum,
                 float* __restrict__ psum, int* __restrict__ tokl,
                 float* __restrict__ gatel) {
  int wid  = threadIdx.x >> 5;
  int lane = threadIdx.x & 31;
  int t = blockIdx.x * (blockDim.x >> 5) + wid;
  if (t >= TT) return;
  const float* xr = x + (size_t)t * MD;

  float sg[NE] = {}, sn[NE] = {};
  for (int i = lane; i < MD; i += 32) {
    float xv = xr[i];
    const float4* wg4 = (const float4*)(Wg + i * NE);
    const float4* wn4 = (const float4*)(Wn + i * NE);
    float4 g0 = wg4[0], g1 = wg4[1], n0 = wn4[0], n1 = wn4[1];
    sg[0] += xv * g0.x; sg[1] += xv * g0.y; sg[2] += xv * g0.z; sg[3] += xv * g0.w;
    sg[4] += xv * g1.x; sg[5] += xv * g1.y; sg[6] += xv * g1.z; sg[7] += xv * g1.w;
    sn[0] += xv * n0.x; sn[1] += xv * n0.y; sn[2] += xv * n0.z; sn[3] += xv * n0.w;
    sn[4] += xv * n1.x; sn[5] += xv * n1.y; sn[6] += xv * n1.z; sn[7] += xv * n1.w;
  }
  #pragma unroll
  for (int off = 16; off > 0; off >>= 1) {
    #pragma unroll
    for (int e = 0; e < NE; ++e) {
      sg[e] += __shfl_xor(sg[e], off, 32);
      sn[e] += __shfl_xor(sn[e], off, 32);
    }
  }
  if (lane != 0) return;

  float logit[NE], ns[NE], H[NE];
  #pragma unroll
  for (int e = 0; e < NE; ++e) {
    logit[e] = sg[e] + bg[e];
    float z = sn[e] + bn[e];
    ns[e] = (z > 20.0f) ? z : log1pf(expf(z));   // softplus
    H[e] = logit[e] + noise[(size_t)t * NE + e] * ns[e];
  }
  float v1 = -1e30f; int i1 = 0;
  #pragma unroll
  for (int e = 0; e < NE; ++e) if (H[e] > v1) { v1 = H[e]; i1 = e; }
  float v2 = -1e30f; int i2 = 0;
  #pragma unroll
  for (int e = 0; e < NE; ++e) if (e != i1 && H[e] > v2) { v2 = H[e]; i2 = e; }
  float v3 = -1e30f;
  #pragma unroll
  for (int e = 0; e < NE; ++e) if (e != i1 && e != i2 && H[e] > v3) v3 = H[e];

  float ex = expf(v2 - v1);
  float s  = 1.0f + ex;
  float g1 = 1.0f / s;
  float g2 = ex / s;

  int p1 = atomicAdd(&cnt[i1], 1);
  tokl[i1 * TT + p1] = t; gatel[i1 * TT + p1] = g1;
  int p2 = atomicAdd(&cnt[i2], 1);
  tokl[i2 * TT + p2] = t; gatel[i2 * TT + p2] = g2;
  atomicAdd(&gsum[i1], g1);
  atomicAdd(&gsum[i2], g2);

  #pragma unroll
  for (int e = 0; e < NE; ++e) {
    float psi = (H[e] > v2) ? v2 : ((H[e] <= v3) ? v3 : H[e]);
    float zz  = (logit[e] - psi) / ns[e];
    atomicAdd(&psum[e], 0.5f * erfcf(-zz * 0.70710678118654752f));
  }
}

// ---------------- bf16 WMMA GEMM with async double-buffered staging ----------------
// C[128x64 tile] = A(bf16,row-major,[rows][lda]) x Bt(bf16,[N][Ktot])^T + bias
template <bool STORE_BF16>
__global__ __launch_bounds__(256)
void gemm_async(const bf16* __restrict__ A, int lda,
                const bf16* __restrict__ Bt, int Ktot,
                const float* __restrict__ bias,
                float* __restrict__ Cf, bf16* __restrict__ Cb, int ldc) {
  __shared__ __align__(16) bf16 As[2 * 128 * LDS_STRIDE];
  __shared__ __align__(16) bf16 Bs[2 * 64 * LDS_STRIDE];

  int tid  = threadIdx.x;
  int lane = tid & 31, wid = tid >> 5;
  int wm = wid & 3, wn = wid >> 2;
  int lmod = lane & 15, lhalf = lane >> 4;
  int k0 = lhalf * 8, moff = lhalf * 8;
  int rowBase = blockIdx.y * 128;
  int colBase = blockIdx.x * 64;

  unsigned ldaB = (unsigned)lda * 2u;
  unsigned ldkB = (unsigned)Ktot * 2u;
  uint64_t aG = (uint64_t)(uintptr_t)A;
  uint64_t bG = (uint64_t)(uintptr_t)Bt;
  unsigned asB = (unsigned)(uintptr_t)&As[0];
  unsigned bsB = (unsigned)(uintptr_t)&Bs[0];

  int ci  = tid & 3;        // 16B chunk within 64B row
  int arA = tid >> 2;       // A rows 0..63 (+64 on 2nd issue)
  int brB = tid >> 2;       // B rows 0..63

  auto stage = [&](int buf, int kk) {
    unsigned kB = (unsigned)kk * 2u + (unsigned)ci * 16u;
    #pragma unroll
    for (int j = 0; j < 2; ++j) {
      int r = arA + j * 64;
      unsigned lds = asB + (unsigned)buf * A_BUF_B + (unsigned)r * (LDS_STRIDE * 2) + ci * 16u;
      unsigned gof = (unsigned)(rowBase + r) * ldaB + kB;
      async_cp16(lds, gof, aG);
    }
    unsigned lds = bsB + (unsigned)buf * B_BUF_B + (unsigned)brB * (LDS_STRIDE * 2) + ci * 16u;
    unsigned gof = (unsigned)(colBase + brB) * ldkB + kB;
    async_cp16(lds, gof, bG);
  };

  v8f z = {0,0,0,0,0,0,0,0};
  v8f acc00 = z, acc01 = z, acc10 = z, acc11 = z;

  stage(0, 0);
  int nK = Ktot / 32;
  for (int k = 0; k < nK; ++k) {
    __syncthreads();                       // all waves done reading buf[(k+1)&1]
    if (k + 1 < nK) { stage((k + 1) & 1, (k + 1) * 32); wait_async3(); }
    else            { wait_async0(); }
    __syncthreads();                       // all waves' stage-k copies landed
    const bf16* Ab = &As[(k & 1) * 128 * LDS_STRIDE];
    const bf16* Bb = &Bs[(k & 1) * 64 * LDS_STRIDE];
    v16bf a0 = ld_frag(Ab, wm * 32,      lmod, k0);
    v16bf a1 = ld_frag(Ab, wm * 32 + 16, lmod, k0);
    v16bf b0 = ld_frag(Bb, wn * 32,      lmod, k0);
    v16bf b1 = ld_frag(Bb, wn * 32 + 16, lmod, k0);
    acc00 = __builtin_amdgcn_wmma_f32_16x16x32_bf16(false, a0, false, b0, (short)0, acc00, false, false);
    acc01 = __builtin_amdgcn_wmma_f32_16x16x32_bf16(false, a0, false, b1, (short)0, acc01, false, false);
    acc10 = __builtin_amdgcn_wmma_f32_16x16x32_bf16(false, a1, false, b0, (short)0, acc10, false, false);
    acc11 = __builtin_amdgcn_wmma_f32_16x16x32_bf16(false, a1, false, b1, (short)0, acc11, false, false);
  }

  #pragma unroll
  for (int fm = 0; fm < 2; ++fm) {
    #pragma unroll
    for (int fn = 0; fn < 2; ++fn) {
      Acc8 u;
      u.v = (fm == 0) ? ((fn == 0) ? acc00 : acc01) : ((fn == 0) ? acc10 : acc11);
      int c = colBase + wn * 32 + fn * 16 + lmod;
      float bv = bias[c];
      #pragma unroll
      for (int i = 0; i < 8; ++i) {
        int r = rowBase + wm * 32 + fm * 16 + moff + i;
        float val = u.f[i] + bv;
        if (STORE_BF16) Cb[(size_t)r * ldc + c] = (bf16)val;
        else            Cf[(size_t)r * ldc + c] = val;
      }
    }
  }
}

// ---------------- expert-gathered GEMM + SiLU*xV*gate epilogue ----------------
__global__ __launch_bounds__(256)
void expert_kernel(const bf16* __restrict__ xb,
                   const bf16* __restrict__ Wet,   // [e][DH][MD] transposed bf16
                   const float* __restrict__ be,
                   const bf16* __restrict__ xV,
                   const int* __restrict__ cnt,
                   const int* __restrict__ tokl, const float* __restrict__ gatel,
                   float* __restrict__ h) {
  __shared__ __align__(16) bf16 As[2 * 128 * LDS_STRIDE];
  __shared__ __align__(16) bf16 Bs[2 * 64 * LDS_STRIDE];
  __shared__ int   sTok[128];
  __shared__ float sGate[128];

  int e  = blockIdx.z;
  int nC = cnt[e];
  if ((int)blockIdx.y * 128 >= nC) return;   // uniform exit

  int tid = threadIdx.x;
  if (tid < 128) {
    int li = blockIdx.y * 128 + tid;
    int t = -1; float g = 0.0f;
    if (li < nC) { t = tokl[e * TT + li]; g = gatel[e * TT + li]; }
    sTok[tid] = t; sGate[tid] = g;
  }
  __syncthreads();

  int lane = tid & 31, wid = tid >> 5;
  int wm = wid & 3, wn = wid >> 2;
  int lmod = lane & 15, lhalf = lane >> 4;
  int k0 = lhalf * 8, moff = lhalf * 8;
  int colBase = blockIdx.x * 64;

  uint64_t aG = (uint64_t)(uintptr_t)xb;
  uint64_t bG = (uint64_t)(uintptr_t)(Wet + (size_t)e * MD * DH);
  unsigned asB = (unsigned)(uintptr_t)&As[0];
  unsigned bsB = (unsigned)(uintptr_t)&Bs[0];

  int ci  = tid & 3;
  int arA = tid >> 2;
  int brB = tid >> 2;

  auto stage = [&](int buf, int kk) {
    unsigned kB = (unsigned)kk * 2u + (unsigned)ci * 16u;
    #pragma unroll
    for (int j = 0; j < 2; ++j) {
      int r = arA + j * 64;
      int t = sTok[r];
      unsigned trow = (unsigned)(t < 0 ? 0 : t);       // junk row -> dropped in epilogue
      unsigned lds = asB + (unsigned)buf * A_BUF_B + (unsigned)r * (LDS_STRIDE * 2) + ci * 16u;
      unsigned gof = trow * (unsigned)(MD * 2) + kB;
      async_cp16(lds, gof, aG);
    }
    unsigned lds = bsB + (unsigned)buf * B_BUF_B + (unsigned)brB * (LDS_STRIDE * 2) + ci * 16u;
    unsigned gof = (unsigned)(colBase + brB) * (unsigned)(MD * 2) + kB;
    async_cp16(lds, gof, bG);
  };

  v8f z = {0,0,0,0,0,0,0,0};
  v8f acc00 = z, acc01 = z, acc10 = z, acc11 = z;

  stage(0, 0);
  constexpr int nK = MD / 32;
  for (int k = 0; k < nK; ++k) {
    __syncthreads();
    if (k + 1 < nK) { stage((k + 1) & 1, (k + 1) * 32); wait_async3(); }
    else            { wait_async0(); }
    __syncthreads();
    const bf16* Ab = &As[(k & 1) * 128 * LDS_STRIDE];
    const bf16* Bb = &Bs[(k & 1) * 64 * LDS_STRIDE];
    v16bf a0 = ld_frag(Ab, wm * 32,      lmod, k0);
    v16bf a1 = ld_frag(Ab, wm * 32 + 16, lmod, k0);
    v16bf b0 = ld_frag(Bb, wn * 32,      lmod, k0);
    v16bf b1 = ld_frag(Bb, wn * 32 + 16, lmod, k0);
    acc00 = __builtin_amdgcn_wmma_f32_16x16x32_bf16(false, a0, false, b0, (short)0, acc00, false, false);
    acc01 = __builtin_amdgcn_wmma_f32_16x16x32_bf16(false, a0, false, b1, (short)0, acc01, false, false);
    acc10 = __builtin_amdgcn_wmma_f32_16x16x32_bf16(false, a1, false, b0, (short)0, acc10, false, false);
    acc11 = __builtin_amdgcn_wmma_f32_16x16x32_bf16(false, a1, false, b1, (short)0, acc11, false, false);
  }

  #pragma unroll
  for (int fm = 0; fm < 2; ++fm) {
    #pragma unroll
    for (int fn = 0; fn < 2; ++fn) {
      Acc8 u;
      u.v = (fm == 0) ? ((fn == 0) ? acc00 : acc01) : ((fn == 0) ? acc10 : acc11);
      int c = colBase + wn * 32 + fn * 16 + lmod;
      float bev = be[(size_t)e * DH + c];
      #pragma unroll
      for (int i = 0; i < 8; ++i) {
        int rl = wm * 32 + fm * 16 + moff + i;
        int t = sTok[rl];
        if (t < 0) continue;
        float w   = u.f[i] + bev;
        float sil = w / (1.0f + expf(-w));                 // SiLU
        float xv  = (float)xV[(size_t)t * DH + c];
        atomicAdd(&h[(size_t)t * DH + c], sGate[rl] * sil * xv);
      }
    }
  }
}

// ---------------- loss finalize ----------------
__global__ void loss_kernel(const float* __restrict__ gsum,
                            const float* __restrict__ psum,
                            float* __restrict__ out) {
  if (threadIdx.x == 0 && blockIdx.x == 0) {
    float mg = 0.f, mp = 0.f;
    for (int e = 0; e < NE; ++e) { mg += gsum[e]; mp += psum[e]; }
    mg /= (float)NE; mp /= (float)NE;
    float vg = 0.f, vp = 0.f;
    for (int e = 0; e < NE; ++e) {
      float d1 = gsum[e] - mg; vg += d1 * d1;
      float d2 = psum[e] - mp; vp += d2 * d2;
    }
    vg /= (float)NE; vp /= (float)NE;
    float L = 0.01f * sqrtf(vg) / (mg + 1e-6f) + 0.01f * sqrtf(vp) / (mp + 1e-6f);
    out[(size_t)TT * MD] = L;
  }
}

// ---------------- host launch ----------------
extern "C" void kernel_launch(void* const* d_in, const int* in_sizes, int n_in,
                              void* d_out, int out_size, void* d_ws, size_t ws_size,
                              hipStream_t stream) {
  const float* x     = (const float*)d_in[0];
  const float* noise = (const float*)d_in[1];
  const float* Wg    = (const float*)d_in[2];
  const float* bg    = (const float*)d_in[3];
  const float* Wn    = (const float*)d_in[4];
  const float* bn    = (const float*)d_in[5];
  const float* We    = (const float*)d_in[6];
  const float* be    = (const float*)d_in[7];
  const float* V     = (const float*)d_in[8];
  const float* bV    = (const float*)d_in[9];
  const float* W2    = (const float*)d_in[10];
  const float* b2    = (const float*)d_in[11];
  float* y = (float*)d_out;

  char* w = (char*)d_ws;
  size_t off = 0;
  auto alloc = [&](size_t bytes) {
    void* p = w + off;
    off = (off + bytes + 255) & ~(size_t)255;
    return p;
  };
  float* h    = (float*)alloc((size_t)TT * DH * sizeof(float));   // f32 atomic accumulator
  bf16*  hb   = (bf16*) alloc((size_t)TT * DH * sizeof(bf16));
  bf16*  xb   = (bf16*) alloc((size_t)TT * MD * sizeof(bf16));
  bf16*  xVb  = (bf16*) alloc((size_t)TT * DH * sizeof(bf16));
  bf16*  Vt   = (bf16*) alloc((size_t)MD * DH * sizeof(bf16));    // [DH][MD]
  bf16*  Wet  = (bf16*) alloc((size_t)NE * MD * DH * sizeof(bf16)); // [e][DH][MD]
  bf16*  W2t  = (bf16*) alloc((size_t)DH * MD * sizeof(bf16));    // [MD][DH]
  int*   cnt  = (int*)  alloc(NE * sizeof(int));
  float* gsum = (float*)alloc(NE * sizeof(float));
  float* psum = (float*)alloc(NE * sizeof(float));
  int*   tokl = (int*)  alloc((size_t)NE * TT * sizeof(int));
  float* gatl = (float*)alloc((size_t)NE * TT * sizeof(float));
  (void)ws_size; (void)in_sizes; (void)n_in; (void)out_size;

  zero_kernel<<<2048, 256, 0, stream>>>(h, (size_t)TT * DH, cnt, gsum, psum);
  // precision prep (bandwidth-trivial, L2-resident weights)
  cvt_bf16_kernel<<<1024, 256, 0, stream>>>(x, xb, (size_t)TT * MD / 4);
  transpose_cvt_kernel<<<dim3(MD / 32, DH / 32, 1), 256, 0, stream>>>(V, Vt, MD, DH);
  transpose_cvt_kernel<<<dim3(MD / 32, DH / 32, NE), 256, 0, stream>>>(We, Wet, MD, DH);
  transpose_cvt_kernel<<<dim3(DH / 32, MD / 32, 1), 256, 0, stream>>>(W2, W2t, DH, MD);

  gate_kernel<<<TT / 8, 256, 0, stream>>>(x, noise, Wg, bg, Wn, bn,
                                          cnt, gsum, psum, tokl, gatl);
  // xV = x @ V + bV  (bf16 out)
  gemm_async<true><<<dim3(DH / 64, TT / 128), 256, 0, stream>>>(
      xb, MD, Vt, MD, bV, nullptr, xVb, DH);
  // per-expert gathered GEMM, SiLU * xV * gate -> atomicAdd into h
  expert_kernel<<<dim3(DH / 64, TT / 128, NE), 256, 0, stream>>>(
      xb, Wet, be, xVb, cnt, tokl, gatl, h);
  // h -> bf16
  cvt_bf16_kernel<<<1024, 256, 0, stream>>>(h, hb, (size_t)TT * DH / 4);
  // y = h @ W2 + b2
  gemm_async<false><<<dim3(MD / 64, TT / 128), 256, 0, stream>>>(
      hb, DH, W2t, DH, b2, y, nullptr, MD);
  loss_kernel<<<1, 32, 0, stream>>>(gsum, psum, y);
}
